// BiFNOk_16020228014757
// MI455X (gfx1250) — matrix operand order repaired and checked by reading, not compiled, not run on hardware
//
#include <hip/hip_runtime.h>
#include <math.h>

// ---------------------------------------------------------------------------
// BiFNO forward for MI455X (gfx1250, wave32).
//  - conv MLPs: LDS-staged implicit GEMM on v_wmma_f32_16x16x32_f16 (f32 accum)
//    (all fragments = contiguous ds_load_b128; epilogue branch-free)
//  - spectral 96x96 complex per-mode mixing: HBM-streaming kernel (+prefetch)
//  - rfft2/irfft2 truncated to 32 modes: factored dense DFT with twiddle tables
// ---------------------------------------------------------------------------

typedef __attribute__((ext_vector_type(16))) _Float16 v16h;
typedef __attribute__((ext_vector_type(8)))  float    v8f;

#define L_LAYERS 4
#define MMODES   32
#define PA 48
#define PB 48
#define PTOT 96
#define HA 256
#define HB 128
#define NPIXA (HA*HA)
#define NPIXB (HB*HB)
#define NMODE (MMODES*MMODES)   // 1024
#define HBP 130                  // padded height/width for branch B (halo = 1)
#define NPIXBP (HBP*HBP)         // 16900
#define KB3 448                  // 48*9 = 432 rounded up to 14 chunks of 32

// gelu (tanh approx) in sigmoid form: 0.5*(1+tanh(t)) == 1/(1+exp(-2t)).
// v_exp_f32 + v_rcp_f32, fully straight-line.
__device__ __forceinline__ float gelu_tanh(float x) {
    const float two_k0 = 1.5957691216057308f;   // 2*sqrt(2/pi)
    const float k1 = 0.044715f;
    float t = two_k0 * (x + k1 * x * x * x);
    return x * __builtin_amdgcn_rcpf(1.0f + __expf(-t));
}

// K index held by element j of a v16h fragment (16-bit operands, wave32).
// A matrix (16x32, MxK): per ISA 7.12.2 table (interleaved K halves).
__device__ __forceinline__ int wmma_ka(int lane, int j) {
    return j + ((j & 8) ? 8 : 0) + ((lane & 16) ? 8 : 0);
}

// ---------------------------------------------------------------------------
// Twiddle tables.
//  colC/colS: [N][32]  cos/sin(2*pi*k*x/N)           (forward cols & inverse cols)
//  rowC/rowS: [32][N]  cos/sin(2*pi*(r-16)*y/N)      (forward rows & inverse rows)
// ---------------------------------------------------------------------------
__global__ void k_twiddle(float* __restrict__ colC, float* __restrict__ colS,
                          float* __restrict__ rowC, float* __restrict__ rowS, int N) {
    const float twopi = 6.283185307179586f;
    int tid = blockIdx.x * blockDim.x + threadIdx.x;
    int half = N * MMODES;
    if (tid < half) {
        int x = tid / MMODES, k = tid % MMODES;
        int ph = (k * x) % N;
        float th = twopi * (float)ph / (float)N;
        colC[tid] = cosf(th);
        colS[tid] = sinf(th);
    } else if (tid < 2*half) {
        int t = tid - half;
        int r = t / N, y = t % N;
        int f = r - MMODES/2;
        int ph = ((f * y) % N + N) % N;
        float th = twopi * (float)ph / (float)N;
        rowC[t] = cosf(th);
        rowS[t] = sinf(th);
    }
}

// Encoder: dst[c,p] = b[c] + w[c,0]*x0[p] + w[c,1]*x1[p] + w[c,2]*u[p]
__global__ void k_encoder(const float* __restrict__ x, const float* __restrict__ u,
                          const float* __restrict__ w, const float* __restrict__ b,
                          float* __restrict__ dst, int nbits) {
    int gid = blockIdx.x * blockDim.x + threadIdx.x;
    int npix = 1 << nbits;
    int c = gid >> nbits;
    int p = gid & (npix - 1);
    float v = b[c] + w[c*3+0]*x[p] + w[c*3+1]*x[npix + p] + w[c*3+2]*u[p];
    dst[(size_t)c*npix + p] = v;
}

// Decoder: dst[p] = b + sum_c w[c]*src[c,p]   (Cout = 1)
__global__ void k_decoder(const float* __restrict__ src, const float* __restrict__ w,
                          const float* __restrict__ b, float* __restrict__ dst, int npix) {
    int p = blockIdx.x * blockDim.x + threadIdx.x;
    float acc = b[0];
    #pragma unroll 4
    for (int c = 0; c < 48; ++c) acc += w[c] * src[(size_t)c*npix + p];
    dst[p] = acc;
}

// Zero-halo pad for branch B: (48,128,128) -> (48,130,130)
__global__ void k_pad130(const float* __restrict__ src, float* __restrict__ dst) {
    int gid = blockIdx.x * blockDim.x + threadIdx.x;
    if (gid >= PB * NPIXBP) return;
    int c = gid / NPIXBP;
    int r = gid - c * NPIXBP;
    int y = r / HBP;
    int x = r - y * HBP;
    float v = 0.0f;
    if (y >= 1 && y <= HB && x >= 1 && x <= HB)
        v = src[(size_t)c*NPIXB + (y-1)*HB + (x-1)];
    dst[gid] = v;
}

// Forward DFT stage 1 (columns): t[c,y,k] = (1/N) * sum_x u[c,y,x] * e^{-2pi i k x/N}
__global__ void k_fdft_cols(const float* __restrict__ u,
                            const float* __restrict__ colC, const float* __restrict__ colS,
                            float* __restrict__ tre, float* __restrict__ tim, int N) {
    int gid = blockIdx.x * blockDim.x + threadIdx.x;   // C*N*32 threads
    int k  = gid & 31;
    int cy = gid >> 5;
    const float* row = u + (size_t)cy * N;
    float ar = 0.f, ai = 0.f;
    for (int x = 0; x < N; ++x) {
        float uv = row[x];
        ar += uv * colC[x*32 + k];
        ai -= uv * colS[x*32 + k];
    }
    float inv = 1.0f / (float)N;
    tre[gid] = ar * inv;
    tim[gid] = ai * inv;
}

// Forward DFT stage 2 (rows, rolled freqs r-16):
// coeff[cofs+c, r, k] = (1/N) * sum_y t[c,y,k] * e^{-2pi i (r-16) y/N}
__global__ void k_fdft_rows(const float* __restrict__ tre, const float* __restrict__ tim,
                            const float* __restrict__ rowC, const float* __restrict__ rowS,
                            float* __restrict__ cre, float* __restrict__ cim,
                            int N, int cofs) {
    int gid = blockIdx.x * blockDim.x + threadIdx.x;   // C*1024 threads
    int k = gid & 31;
    int r = (gid >> 5) & 31;
    int c = gid >> 10;
    float ar = 0.f, ai = 0.f;
    for (int y = 0; y < N; ++y) {
        float tr = tre[((size_t)c*N + y)*32 + k];
        float ti = tim[((size_t)c*N + y)*32 + k];
        float cs = rowC[r*N + y];
        float sn = rowS[r*N + y];
        ar += tr*cs + ti*sn;
        ai += ti*cs - tr*sn;
    }
    float inv = 1.0f / (float)N;
    size_t o = (size_t)(cofs + c)*NMODE + r*32 + k;
    cre[o] = ar * inv;
    cim[o] = ai * inv;
}

// Spectral mixing: y[p,mn] = sum_q (Wr[p,q,mn] + i Wi[p,q,mn]) * (xr[q,mn] + i xi[q,mn])
// Pure HBM stream over 75.5 MB of weights per layer; coalesced over mn.
__global__ void k_spectral(const float* __restrict__ Wr, const float* __restrict__ Wi,
                           const float* __restrict__ xre, const float* __restrict__ xim,
                           float* __restrict__ yre, float* __restrict__ yim) {
    int mn = blockIdx.x * blockDim.x + threadIdx.x;    // 0..1023
    int p  = blockIdx.y;                               // 0..95
    const float* wr = Wr + (size_t)p * PTOT * NMODE;
    const float* wi = Wi + (size_t)p * PTOT * NMODE;
    float ar = 0.f, ai = 0.f;
    for (int q = 0; q < PTOT; ++q) {
        if (q + 8 < PTOT) {
            __builtin_prefetch(wr + (size_t)(q + 8)*NMODE + mn, 0, 1);
            __builtin_prefetch(wi + (size_t)(q + 8)*NMODE + mn, 0, 1);
        }
        float wrv = wr[(size_t)q*NMODE + mn];
        float wiv = wi[(size_t)q*NMODE + mn];
        float xr  = xre[(size_t)q*NMODE + mn];
        float xi  = xim[(size_t)q*NMODE + mn];
        ar += wrv*xr - wiv*xi;
        ai += wrv*xi + wiv*xr;
    }
    yre[(size_t)p*NMODE + mn] = ar;
    yim[(size_t)p*NMODE + mn] = ai;
}

// Inverse stage 1 (rows): s[c,y,k] = sum_r coeff[cofs+c,r,k] * e^{+2pi i (r-16) y/N}
__global__ void k_idft_rows(const float* __restrict__ cre, const float* __restrict__ cim,
                            const float* __restrict__ rowC, const float* __restrict__ rowS,
                            float* __restrict__ sre, float* __restrict__ sim,
                            int N, int cofs) {
    int gid = blockIdx.x * blockDim.x + threadIdx.x;   // C*N*32 threads
    int k = gid & 31;
    int rest = gid >> 5;
    int y = rest % N;      // N is a power of two; cheap
    int c = rest / N;
    float ar = 0.f, ai = 0.f;
    #pragma unroll 4
    for (int r = 0; r < 32; ++r) {
        size_t o = (size_t)(cofs + c)*NMODE + r*32 + k;
        float crv = cre[o], civ = cim[o];
        float cs = rowC[r*N + y];
        float sn = rowS[r*N + y];
        ar += crv*cs - civ*sn;
        ai += crv*sn + civ*cs;
    }
    sre[gid] = ar;
    sim[gid] = ai;
}

// Inverse stage 2 (cols, real output):
// u[c,y,x] = sum_k wgt_k * Re(s[c,y,k] * e^{+2pi i k x/N}),  wgt_0=1 else 2
__global__ void k_idft_cols(const float* __restrict__ sre, const float* __restrict__ sim,
                            const float* __restrict__ colC, const float* __restrict__ colS,
                            float* __restrict__ dst, int nbits) {
    int gid = blockIdx.x * blockDim.x + threadIdx.x;   // C*N*N threads
    int N = 1 << nbits;
    int x  = gid & (N - 1);
    int cy = gid >> nbits;
    float acc = 0.f;
    #pragma unroll 4
    for (int k = 0; k < 32; ++k) {
        float w = (k == 0) ? 1.0f : 2.0f;
        float sr = sre[(size_t)cy*32 + k];
        float si = sim[(size_t)cy*32 + k];
        acc += w * (sr * colC[x*32 + k] - si * colS[x*32 + k]);
    }
    dst[gid] = acc;
}

// ---------------------------------------------------------------------------
// 1x1 conv 48->48 over 256x256 as LDS-staged WMMA GEMM.
// Block = 128 threads (4 waves). Block tile: 16 cout x 64 pixels. K padded to 64.
// LDS: W tile [m][64] f16 (2KB) + X tile pixel-major [n][64] f16 (8KB).
// All fragments are contiguous b128 LDS reads. RESID is compile-time.
// ---------------------------------------------------------------------------
template <int RESID>
__global__ void k_conv1x1A_wmma(const float* __restrict__ src,
                                const float* __restrict__ w,    // 48x48
                                const float* __restrict__ bias,
                                const float* __restrict__ res,
                                float* __restrict__ dst) {
    __shared__ alignas(16) _Float16 sW[16*64];     // [m][k]
    __shared__ alignas(16) _Float16 sX[64*64];     // [n][k]  (pixel-major!)
    int tid   = threadIdx.x;
    int lane  = tid & 31;
    int wave  = tid >> 5;
    int pixb0 = blockIdx.x * 64;
    int coutb = blockIdx.y * 16;
    int l15   = lane & 15;
    int hi    = (lane & 16) ? 1 : 0;

    // cooperative staging (coalesced global reads over pixels; transposed LDS write)
    for (int e = tid; e < 16*64; e += 128) {
        int m = e >> 6, k = e & 63;
        sW[e] = (k < 48) ? (_Float16)w[(coutb + m)*48 + k] : (_Float16)0.0f;
    }
    for (int e = tid; e < 64*64; e += 128) {
        int k = e >> 6, n = e & 63;
        sX[n*64 + k] = (k < 48) ? (_Float16)src[(size_t)k*NPIXA + pixb0 + n]
                                : (_Float16)0.0f;
    }
    __syncthreads();

    int nloc = wave*16 + l15;     // this lane's pixel within the block tile
    v8f acc = {};
    #pragma unroll
    for (int kb = 0; kb < 64; kb += 32) {
        v16h a;
        #pragma unroll
        for (int j = 0; j < 16; ++j)
            a[j] = sW[l15*64 + kb + 16*hi + j];
        // B fragment: K = kb + 16*hi + j, j=0..15 contiguous -> single v16h read
        v16h b = *(const v16h*)&sX[nloc*64 + kb + 16*hi];
        acc = __builtin_amdgcn_wmma_f32_16x16x32_f16(false, a, false, b,
                                                     (short)0, acc, false, false);
    }
    int mhi = hi ? 8 : 0;
    int n = pixb0 + nloc;
    #pragma unroll
    for (int i = 0; i < 8; ++i) {
        int m = coutb + mhi + i;
        float v = gelu_tanh(acc[i] + bias[m]);
        if (RESID) v += res[(size_t)m*NPIXA + n];
        dst[(size_t)m*NPIXA + n] = v;
    }
}

// ---------------------------------------------------------------------------
// 3x3 SAME conv 48->48 over 128x128 as LDS-staged implicit-GEMM WMMA.
// src is the zero-halo padded buffer (48,130,130) -> unconditional im2col loads.
// Block = 128 threads (4 waves). Block tile: 16 cout x 64 pixels. K = 432 -> 448.
// LDS: W tile [m][448] f16 (14KB, staged once) + X chunk pixel-major [n][32] f16.
// ---------------------------------------------------------------------------
template <int RESID>
__global__ void k_conv3x3B_wmma(const float* __restrict__ src,  // padded
                                const float* __restrict__ w,    // 48x48x3x3
                                const float* __restrict__ bias,
                                const float* __restrict__ res,
                                float* __restrict__ dst) {
    __shared__ alignas(16) _Float16 sW[16*KB3];    // [m][k]
    __shared__ alignas(16) _Float16 sX[64*32];     // [n][kk] (pixel-major!)
    int tid   = threadIdx.x;
    int lane  = tid & 31;
    int wave  = tid >> 5;
    int pixb0 = blockIdx.x * 64;
    int coutb = blockIdx.y * 16;
    int l15   = lane & 15;
    int hi    = (lane & 16) ? 1 : 0;

    // stage all weights once (K padded with zeros)
    for (int e = tid; e < 16*KB3; e += 128) {
        int m = e / KB3, k = e - m*KB3;
        _Float16 v = (_Float16)0.0f;
        if (k < 432) {
            int ci = k / 9, t = k - ci*9;
            v = (_Float16)w[((coutb + m)*48 + ci)*9 + t];
        }
        sW[e] = v;
    }

    int nloc = wave*16 + l15;
    v8f acc = {};
    for (int kb = 0; kb < KB3; kb += 32) {
        __syncthreads();
        // stage activation chunk, transposed into [n][kk]
        #pragma unroll
        for (int e = tid; e < 32*64; e += 128) {
            int kk = e >> 6, n = e & 63;
            int K = kb + kk;
            _Float16 v = (_Float16)0.0f;
            if (K < 432) {
                int ci = K / 9, t = K - ci*9;
                int dy = t / 3, dx = t - dy*3;          // 0..2 (halo origin -1)
                int pix = pixb0 + n;
                int py = pix >> 7, px = pix & 127;
                v = (_Float16)src[(size_t)ci*NPIXBP + (py + dy)*HBP + (px + dx)];
            }
            sX[n*32 + kk] = v;
        }
        __syncthreads();

        v16h a;
        #pragma unroll
        for (int j = 0; j < 16; ++j)
            a[j] = sW[l15*KB3 + kb + 16*hi + j];
        // B fragment: kk = 16*hi + j contiguous -> single v16h read
        v16h b = *(const v16h*)&sX[nloc*32 + 16*hi];
        acc = __builtin_amdgcn_wmma_f32_16x16x32_f16(false, a, false, b,
                                                     (short)0, acc, false, false);
    }
    int mhi = hi ? 8 : 0;
    int n = pixb0 + nloc;
    #pragma unroll
    for (int i = 0; i < 8; ++i) {
        int m = coutb + mhi + i;
        float v = gelu_tanh(acc[i] + bias[m]);
        if (RESID) v += res[(size_t)m*NPIXB + n];
        dst[(size_t)m*NPIXB + n] = v;
    }
}

// ---------------------------------------------------------------------------
extern "C" void kernel_launch(void* const* d_in, const int* in_sizes, int n_in,
                              void* d_out, int out_size, void* d_ws, size_t ws_size,
                              hipStream_t stream) {
    const float* u_a     = (const float*)d_in[0];
    const float* x_a     = (const float*)d_in[1];
    const float* u_b     = (const float*)d_in[2];
    const float* x_b     = (const float*)d_in[3];
    const float* enc_a_w = (const float*)d_in[4];
    const float* enc_a_b = (const float*)d_in[5];
    const float* enc_b_w = (const float*)d_in[6];
    const float* enc_b_b = (const float*)d_in[7];
    const float* dec_a_w = (const float*)d_in[8];
    const float* dec_a_b = (const float*)d_in[9];
    const float* dec_b_w = (const float*)d_in[10];
    const float* dec_b_b = (const float*)d_in[11];
    const float* c1a_w   = (const float*)d_in[12];
    const float* c1a_b   = (const float*)d_in[13];
    const float* c2a_w   = (const float*)d_in[14];
    const float* c2a_b   = (const float*)d_in[15];
    const float* c1b_w   = (const float*)d_in[16];
    const float* c1b_b   = (const float*)d_in[17];
    const float* c2b_w   = (const float*)d_in[18];
    const float* c2b_b   = (const float*)d_in[19];
    const float* A_re    = (const float*)d_in[20];
    const float* A_im    = (const float*)d_in[21];

    float* ws = (float*)d_ws;
    size_t off = 0;
    auto alloc = [&](size_t n) { float* p = ws + off; off += n; return p; };
    float* twAcC = alloc(HA*32); float* twAcS = alloc(HA*32);
    float* twArC = alloc(32*HA); float* twArS = alloc(32*HA);
    float* twBcC = alloc(HB*32); float* twBcS = alloc(HB*32);
    float* twBrC = alloc(32*HB); float* twBrS = alloc(32*HB);
    float* ua    = alloc((size_t)PA*NPIXA);
    float* ub    = alloc((size_t)PB*NPIXB);
    float* uas   = alloc((size_t)PA*NPIXA);   // post-spectral activations
    float* ubs   = alloc((size_t)PB*NPIXB);
    float* hidA  = alloc((size_t)PA*NPIXA);
    float* hidB  = alloc((size_t)PB*NPIXB);
    float* ubpad = alloc((size_t)PB*NPIXBP);  // halo-padded conv3x3 input
    float* tAre = alloc((size_t)PA*HA*32); float* tAim = alloc((size_t)PA*HA*32);
    float* tBre = alloc((size_t)PB*HB*32); float* tBim = alloc((size_t)PB*HB*32);
    float* c1re = alloc((size_t)PTOT*NMODE); float* c1im = alloc((size_t)PTOT*NMODE);
    float* c2re = alloc((size_t)PTOT*NMODE); float* c2im = alloc((size_t)PTOT*NMODE);
    // t-buffers are dead after k_fdft_rows; reuse them for the inverse stage-1 output
    float* sAre = tAre; float* sAim = tAim;
    float* sBre = tBre; float* sBim = tBim;

    k_twiddle<<<(2*HA*32)/256, 256, 0, stream>>>(twAcC, twAcS, twArC, twArS, HA);
    k_twiddle<<<(2*HB*32)/256, 256, 0, stream>>>(twBcC, twBcS, twBrC, twBrS, HB);

    k_encoder<<<(PA*NPIXA)/256, 256, 0, stream>>>(x_a, u_a, enc_a_w, enc_a_b, ua, 16);
    k_encoder<<<(PB*NPIXB)/256, 256, 0, stream>>>(x_b, u_b, enc_b_w, enc_b_b, ub, 14);

    int padBlocks = (PB*NPIXBP + 255) / 256;
    for (int i = 0; i < L_LAYERS; ++i) {
        // forward truncated DFT -> coeff (96,32,32) complex
        k_fdft_cols<<<(PA*HA*32)/256, 256, 0, stream>>>(ua, twAcC, twAcS, tAre, tAim, HA);
        k_fdft_cols<<<(PB*HB*32)/256, 256, 0, stream>>>(ub, twBcC, twBcS, tBre, tBim, HB);
        k_fdft_rows<<<(PA*NMODE)/256, 256, 0, stream>>>(tAre, tAim, twArC, twArS,
                                                        c1re, c1im, HA, 0);
        k_fdft_rows<<<(PB*NMODE)/256, 256, 0, stream>>>(tBre, tBim, twBrC, twBrS,
                                                        c1re, c1im, HB, PA);
        // per-mode 96x96 complex mixing (bandwidth-bound weight stream)
        dim3 sg(NMODE/256, PTOT);
        k_spectral<<<sg, 256, 0, stream>>>(A_re + (size_t)i*PTOT*PTOT*NMODE,
                                           A_im + (size_t)i*PTOT*PTOT*NMODE,
                                           c1re, c1im, c2re, c2im);
        // inverse DFT -> spatial activations
        k_idft_rows<<<(PA*HA*32)/256, 256, 0, stream>>>(c2re, c2im, twArC, twArS,
                                                        sAre, sAim, HA, 0);
        k_idft_rows<<<(PB*HB*32)/256, 256, 0, stream>>>(c2re, c2im, twBrC, twBrS,
                                                        sBre, sBim, HB, PA);
        k_idft_cols<<<(PA*NPIXA)/256, 256, 0, stream>>>(sAre, sAim, twAcC, twAcS, uas, 8);
        k_idft_cols<<<(PB*NPIXB)/256, 256, 0, stream>>>(sBre, sBim, twBcC, twBcS, ubs, 7);

        // branch A: ua = ua + gelu(conv2(gelu(conv1(uas))))   [LDS-staged WMMA]
        dim3 ga(NPIXA/64, 3);
        k_conv1x1A_wmma<0><<<ga, 128, 0, stream>>>(uas, c1a_w + i*PA*PA, c1a_b + i*PA,
                                                   nullptr, hidA);
        k_conv1x1A_wmma<1><<<ga, 128, 0, stream>>>(hidA, c2a_w + i*PA*PA, c2a_b + i*PA,
                                                   ua, ua);
        // branch B: ub = ub + gelu(conv3x3(gelu(conv3x3(ubs))))  [pad + LDS WMMA]
        dim3 gb(NPIXB/64, 3);
        k_pad130<<<padBlocks, 256, 0, stream>>>(ubs, ubpad);
        k_conv3x3B_wmma<0><<<gb, 128, 0, stream>>>(ubpad, c1b_w + i*PB*PB*9,
                                                   c1b_b + i*PB, nullptr, hidB);
        k_pad130<<<padBlocks, 256, 0, stream>>>(hidB, ubpad);
        k_conv3x3B_wmma<1><<<gb, 128, 0, stream>>>(ubpad, c2b_w + i*PB*PB*9,
                                                   c2b_b + i*PB, ub, ub);
    }

    float* outA = (float*)d_out;
    float* outB = outA + NPIXA;
    k_decoder<<<NPIXA/256, 256, 0, stream>>>(ua, dec_a_w, dec_a_b, outA, NPIXA);
    k_decoder<<<NPIXB/256, 256, 0, stream>>>(ub, dec_b_w, dec_b_b, outB, NPIXB);
}
